// MOGCN_66503273611466
// MI455X (gfx1250) — 8-bit matrix operand order repaired and checked by reading, compiled, not run on hardware
//
#include <hip/hip_runtime.h>
#include <stdint.h>

#define B_    16
#define N_    1024
#define IND   512
#define HIDD  256
#define KL    3
#define NTYP  3
#define ALPHA 0.2f
#define NEGV  -9.0e15f

typedef __attribute__((ext_vector_type(2))) float v2f;
typedef __attribute__((ext_vector_type(8))) float v8f;
typedef __attribute__((ext_vector_type(8))) int   v8i;

static __device__ __forceinline__ v8f wmma_f32(v2f a, v2f b, v8f c) {
  return __builtin_amdgcn_wmma_f32_16x16x4_f32(false, a, false, b, (short)0, c, false, false);
}
static __device__ __forceinline__ v8i wmma_u8(v8i a, v8i b, v8i c) {
  return __builtin_amdgcn_wmma_i32_16x16x64_iu8(false, a, false, b, c, false, false);
}

__global__ void k_binadj(const float* __restrict__ adj, uint8_t* __restrict__ adjb, long total4) {
  long x = (long)blockIdx.x * blockDim.x + threadIdx.x;
  if (x >= total4) return;
  const float4 v = ((const float4*)adj)[x];
  uint32_t p = (v.x > 0.f ? 1u : 0u)
             | ((v.y > 0.f ? 1u : 0u) << 8)
             | ((v.z > 0.f ? 1u : 0u) << 16)
             | ((v.w > 0.f ? 1u : 0u) << 24);
  ((uint32_t*)adjb)[x] = p;
}

__global__ void k_power(const uint8_t* __restrict__ Pin, const uint8_t* __restrict__ Adj,
                        uint8_t* __restrict__ Pout) {
  const int i0 = blockIdx.x * 16, j0 = blockIdx.y * 16;
  const size_t base = (size_t)blockIdx.z * N_ * N_;
  const uint8_t* A  = Pin + base;
  const uint8_t* Bm = Adj + base;
  uint8_t*       C  = Pout + base;
  const int lane = threadIdx.x;
  const int ml   = lane & 15;
  const int hi   = (lane >= 16) ? 1 : 0;

  v8i acc = {0, 0, 0, 0, 0, 0, 0, 0};
  const uint8_t* arow = A + (size_t)(i0 + ml) * N_ + (hi ? 8 : 0);
  const uint8_t* bcol = Bm + j0 + ml;

  for (int k0 = 0; k0 < N_; k0 += 64) {
    v8i a;
    {
      const int2 c0 = *(const int2*)(arow + k0 + 0);
      const int2 c1 = *(const int2*)(arow + k0 + 16);
      const int2 c2 = *(const int2*)(arow + k0 + 32);
      const int2 c3 = *(const int2*)(arow + k0 + 48);
      a[0] = c0.x; a[1] = c0.y; a[2] = c1.x; a[3] = c1.y;
      a[4] = c2.x; a[5] = c2.y; a[6] = c3.x; a[7] = c3.y;
    }
    v8i b;
#pragma unroll
    for (int v = 0; v < 8; ++v) {
      const int kb = k0 + 16 * (v >> 1) + 4 * (v & 1) + (hi ? 8 : 0);
      uint32_t p =  (uint32_t)bcol[(size_t)(kb + 0) * N_]
                 | ((uint32_t)bcol[(size_t)(kb + 1) * N_] << 8)
                 | ((uint32_t)bcol[(size_t)(kb + 2) * N_] << 16)
                 | ((uint32_t)bcol[(size_t)(kb + 3) * N_] << 24);
      b[v] = (int)p;
    }
    acc = wmma_u8(a, b, acc);
  }
  const int rb = hi ? 8 : 0;
#pragma unroll
  for (int v = 0; v < 8; ++v)
    C[(size_t)(i0 + rb + v) * N_ + j0 + ml] = acc[v] ? (uint8_t)1 : (uint8_t)0;
}

__global__ void k_xw(const float* __restrict__ X, const float* __restrict__ Wp,
                     float* __restrict__ h) {
  const int r0 = blockIdx.x * 16;
  const int c0 = blockIdx.y * 16;
  const int lane = threadIdx.x;
  const int ml = lane & 15;
  const int kb = (lane < 16) ? 0 : 2;
  v8f acc = {};
  const float* arow = X + (size_t)(r0 + ml) * IND;
  for (int k0 = 0; k0 < IND; k0 += 4) {
    v2f a = *(const v2f*)(arow + k0 + kb);
    v2f b;
    b.x = Wp[(size_t)(k0 + kb) * HIDD + c0 + ml];
    b.y = Wp[(size_t)(k0 + kb + 1) * HIDD + c0 + ml];
    acc = wmma_f32(a, b, acc);
  }
  const int rb = (lane < 16) ? 0 : 8;
#pragma unroll
  for (int v = 0; v < 8; ++v)
    h[(size_t)(r0 + rb + v) * HIDD + c0 + ml] = acc[v];
}

__global__ void k_f1f2(const float* __restrict__ h, const float* __restrict__ a1,
                       const float* __restrict__ a2, float* __restrict__ f1,
                       float* __restrict__ f2) {
  const int r = blockIdx.x * blockDim.x + threadIdx.x;
  if (r >= B_ * N_) return;
  const float* hr = h + (size_t)r * HIDD;
  float s1 = 0.f, s2 = 0.f;
  for (int f = 0; f < HIDD; ++f) { const float hv = hr[f]; s1 += hv * a1[f]; s2 += hv * a2[f]; }
  f1[r] = s1; f2[r] = s2;
}

__global__ __launch_bounds__(512) void k_attn(const uint8_t* __restrict__ P,
                                              const float* __restrict__ h,
                                              const float* __restrict__ f1,
                                              const float* __restrict__ f2,
                                              float* __restrict__ H,
                                              int t, int klayer, int accum) {
  __shared__ __align__(16) float att[16][1026];
  __shared__ float f2s[N_];
  const int b    = blockIdx.y;
  const int i0   = blockIdx.x * 16;
  const int wave = threadIdx.x >> 5;
  const int lane = threadIdx.x & 31;

  for (int j = threadIdx.x; j < N_; j += 512) f2s[j] = f2[(size_t)b * N_ + j];
  __syncthreads();

  {
    const int gi = i0 + wave;
    const float f1v = f1[(size_t)b * N_ + gi];
    const uint8_t* mrow = P + ((size_t)(b * NTYP + t) * N_ + gi) * N_;
    float mx = -3.402823466e38f;
    for (int j = lane; j < N_; j += 32) {
      float e = f1v + f2s[j];
      e = (e > 0.f) ? e : ALPHA * e;
      const float v = mrow[j] ? e : NEGV;
      att[wave][j] = v;
      mx = fmaxf(mx, v);
    }
    for (int o = 16; o > 0; o >>= 1) mx = fmaxf(mx, __shfl_xor(mx, o, 32));
    float s = 0.f;
    for (int j = lane; j < N_; j += 32) {
      const float ev = __expf(att[wave][j] - mx);
      att[wave][j] = ev;
      s += ev;
    }
    for (int o = 16; o > 0; o >>= 1) s += __shfl_xor(s, o, 32);
    const float inv = 1.f / s;
    for (int j = lane; j < N_; j += 32) att[wave][j] *= inv;
  }
  __syncthreads();

  const int c0 = wave * 16;
  const int ml = lane & 15;
  const int kb = (lane < 16) ? 0 : 2;
  v8f acc = {};
  const float* hb = h + (size_t)b * N_ * HIDD;
  for (int k0 = 0; k0 < N_; k0 += 4) {
    v2f a = *(const v2f*)&att[ml][k0 + kb];
    v2f bv;
    bv.x = hb[(size_t)(k0 + kb) * HIDD + c0 + ml];
    bv.y = hb[(size_t)(k0 + kb + 1) * HIDD + c0 + ml];
    acc = wmma_f32(a, bv, acc);
  }
  const int rb = (lane < 16) ? 0 : 8;
#pragma unroll
  for (int v = 0; v < 8; ++v) {
    const size_t o = (((size_t)b * N_ + i0 + rb + v) * KL + klayer) * HIDD + c0 + ml;
    if (accum) H[o] += acc[v]; else H[o] = acc[v];
  }
}

__global__ void k_final(const float* __restrict__ H, const float* __restrict__ Ww,
                        const float* __restrict__ bw, const float* __restrict__ Wc,
                        float* __restrict__ out) {
  const int r = blockIdx.x * blockDim.x + threadIdx.x;
  if (r >= B_ * N_) return;
  const float* Hr = H + (size_t)r * KL * HIDD;
  float s[KL];
  for (int k = 0; k < KL; ++k) {
    const float* Hk = Hr + k * HIDD;
    float acc = 0.f;
    for (int c = 0; c < 100; ++c) {
      float u = bw[c];
      for (int f = 0; f < HIDD; ++f) u += Hk[f] * Ww[f * 100 + c];
      acc += tanhf(u) * Wc[c];
    }
    s[k] = acc;
  }
  float mx = fmaxf(s[0], fmaxf(s[1], s[2]));
  const float e0 = __expf(s[0] - mx), e1 = __expf(s[1] - mx), e2 = __expf(s[2] - mx);
  const float inv = 1.f / (e0 + e1 + e2);
  const float w0 = e0 * inv, w1 = e1 * inv, w2 = e2 * inv;
  float* outr = out + (size_t)r * HIDD;
  for (int f = 0; f < HIDD; ++f)
    outr[f] = w0 * Hr[f] + w1 * Hr[HIDD + f] + w2 * Hr[2 * HIDD + f];
}

extern "C" void kernel_launch(void* const* d_in, const int* in_sizes, int n_in,
                              void* d_out, int out_size, void* d_ws, size_t ws_size,
                              hipStream_t stream) {
  (void)in_sizes; (void)n_in; (void)out_size; (void)ws_size;
  const float* adj = (const float*)d_in[0];
  const float* X   = (const float*)d_in[1];
  const float* W   = (const float*)d_in[2];
  const float* a1  = (const float*)d_in[3];
  const float* a2  = (const float*)d_in[4];
  const float* Ww  = (const float*)d_in[5];
  const float* bw  = (const float*)d_in[6];
  const float* Wc  = (const float*)d_in[7];
  float* out = (float*)d_out;

  char* p = (char*)d_ws;
  const size_t szA = (size_t)B_ * NTYP * N_ * N_;
  uint8_t* adjb = (uint8_t*)p; p += szA;
  uint8_t* P2   = (uint8_t*)p; p += szA;
  uint8_t* P3   = (uint8_t*)p; p += szA;
  float* h  = (float*)p; p += (size_t)B_ * N_ * HIDD * 4;
  float* f1 = (float*)p; p += (size_t)B_ * N_ * 4;
  float* f2 = (float*)p; p += (size_t)B_ * N_ * 4;
  float* Hb = (float*)p; p += (size_t)B_ * N_ * KL * HIDD * 4;

  const long total4 = (long)B_ * NTYP * N_ * N_ / 4;
  k_binadj<<<dim3((unsigned)((total4 + 255) / 256)), dim3(256), 0, stream>>>(adj, adjb, total4);

  uint8_t* Pcur[KL] = {adjb, P2, P3};
  for (int k = 0; k < KL; ++k) {
    for (int t = 0; t < NTYP; ++t) {
      const float* Wtk = W + (size_t)(t * KL + k) * IND * HIDD;
      k_xw<<<dim3(B_ * N_ / 16, HIDD / 16), dim3(32), 0, stream>>>(X, Wtk, h);
      k_f1f2<<<dim3(B_ * N_ / 256), dim3(256), 0, stream>>>(
          h, a1 + (size_t)(t * KL + k) * HIDD, a2 + (size_t)(t * KL + k) * HIDD, f1, f2);
      k_attn<<<dim3(N_ / 16, B_), dim3(512), 0, stream>>>(Pcur[k], h, f1, f2, Hb, t, k, t > 0);
    }
    if (k + 1 < KL) {
      k_power<<<dim3(N_ / 16, N_ / 16, B_ * NTYP), dim3(32), 0, stream>>>(Pcur[k], adjb, Pcur[k + 1]);
    }
  }
  k_final<<<dim3(B_ * N_ / 256), dim3(256), 0, stream>>>(Hb, Ww, bw, Wc, out);
}
